// MultiHeadSelfAttention_42485816492467
// MI455X (gfx1250) — matrix operand order repaired and checked by reading, not compiled, not run on hardware
//
#include <hip/hip_runtime.h>
#include <hip/hip_fp16.h>

typedef __attribute__((ext_vector_type(16))) _Float16 v16h;
typedef __attribute__((ext_vector_type(8)))  _Float16 v8h;
typedef __attribute__((ext_vector_type(8)))  float    v8f;
typedef __attribute__((ext_vector_type(4)))  float    v4f;

#define BATCH 8
#define SEQ   2048
#define EMBED 384
#define HEADS 4
#define HDIM  96
#define NROWS (BATCH * SEQ)   // 16384

static __device__ inline v16h pack16(v8h a, v8h b) {
  v16h r;
#pragma unroll
  for (int i = 0; i < 8; ++i) { r[i] = a[i]; r[8 + i] = b[i]; }
  return r;
}

// ---------------------------------------------------------------------------
// Kernel 1: fused QKV projection.  4 waves/block, one 16x16 tile per wave,
// K=384 in 12 steps of v_wmma_f32_16x16x32_f16.  The 32x16 W chunk is shared
// by all 4 waves -> staged through LDS (transposed) once per block per step,
// so each lane reads its B fragment as two ds_load_b128.  Output f16 in
// head-major [B,H,S,Hd] layout; store addressing is tile-uniform (SALU).
// ---------------------------------------------------------------------------
__global__ __launch_bounds__(128, 1)
void qkv_proj_kernel(const float* __restrict__ x,
                     const float* __restrict__ Wq, const float* __restrict__ bq,
                     const float* __restrict__ Wk, const float* __restrict__ bk,
                     const float* __restrict__ Wv, const float* __restrict__ bv,
                     _Float16* __restrict__ Qh, _Float16* __restrict__ Kh,
                     _Float16* __restrict__ Vh)
{
  __shared__ _Float16 Wl[16 * 32];   // [col][k] transposed chunk
  const int lane = threadIdx.x & 31;
  const int wave = threadIdx.x >> 5;
  const int ln   = lane & 15;
  const int hi   = lane >> 4;
  const int tileM = blockIdx.x * 4 + wave;   // 0..1023
  const int tileN = blockIdx.y;              // 0..23
  const int proj  = blockIdx.z;              // 0=Q 1=K 2=V

  const float* __restrict__ W    = (proj == 0) ? Wq : (proj == 1) ? Wk : Wv;
  const float* __restrict__ bias = (proj == 0) ? bq : (proj == 1) ? bk : bv;
  _Float16* __restrict__ out     = (proj == 0) ? Qh : (proj == 1) ? Kh : Vh;

  const int arow = tileM * 16 + ln;   // A-matrix row for this lane
  const int ccol = tileN * 16 + ln;   // C/D column for this lane

  // cooperative W-staging indices (block-uniform pattern)
  const int scl = threadIdx.x & 15;   // staged column 0..15
  const int sk0 = threadIdx.x >> 4;   // staged k base 0..7

  v8f acc = {};
#pragma unroll 1
  for (int kk = 0; kk < EMBED; kk += 32) {
    __syncthreads();   // previous iteration's B reads are done
    // stage W[kk..kk+31][tileN*16..+15] -> Wl[col][k]  (coalesced over cols)
#pragma unroll
    for (int j = 0; j < 4; ++j) {
      const int kl = sk0 + 8 * j;
      Wl[scl * 32 + kl] =
          (_Float16)W[(size_t)(kk + kl) * EMBED + tileN * 16 + scl];
    }
    __syncthreads();

    // A fragment (16x32 f16): lane holds K = kk+{hi*8..+7} and +16.
    const float* ap = x + (size_t)arow * EMBED + kk + hi * 8;
    v4f a0 = *(const v4f*)(ap);
    v4f a1 = *(const v4f*)(ap + 4);
    v4f a2 = *(const v4f*)(ap + 16);
    v4f a3 = *(const v4f*)(ap + 20);
    v16h a;
#pragma unroll
    for (int i = 0; i < 4; ++i) {
      a[i]      = (_Float16)a0[i];
      a[4 + i]  = (_Float16)a1[i];
      a[8 + i]  = (_Float16)a2[i];
      a[12 + i] = (_Float16)a3[i];
    }
    // B fragment: K = {hi*8..+7} and +16 at this lane's column, from LDS.
    const _Float16* wp = &Wl[ln * 32 + hi * 8];
    v16h b = pack16(*(const v8h*)wp, *(const v8h*)(wp + 16));
    acc = __builtin_amdgcn_wmma_f32_16x16x32_f16(false, a, false, b, (short)0,
                                                 acc, false, false);
  }

  // Store: all divisions are on tile-uniform values -> scalar ops.
  const float bc = bias[ccol];
  const int bb = (tileM * 16) / SEQ;             // batch (tile-uniform)
  const int s0 = tileM * 16 - bb * SEQ;          // seq row base
  const int hh = (tileN * 16) / HDIM;            // head (tile-uniform)
  const int hd = (tileN * 16 - hh * HDIM) + ln;  // hd column
  _Float16* op = out + (((size_t)bb * HEADS + hh) * SEQ + s0 + hi * 8) * HDIM + hd;
#pragma unroll
  for (int r = 0; r < 8; ++r)
    op[(size_t)r * HDIM] = (_Float16)(acc[r] + bc);
}

// ---------------------------------------------------------------------------
// Kernel 2: causal flash attention.  One wave per (b, h, 16-row q tile).
// Per 32-key chunk: 6 WMMAs for scores, online softmax, P staged through LDS
// to A-layout, V chunk staged through LDS (b128 row writes, 16-bit column reads), then
// 6 WMMAs for PV.  Same-wave DS ops are in-order -> no barriers needed.
// ---------------------------------------------------------------------------
__global__ __launch_bounds__(32, 1)
void flash_attn_kernel(const _Float16* __restrict__ Qh,
                       const _Float16* __restrict__ Kh,
                       const _Float16* __restrict__ Vh,
                       _Float16* __restrict__ attn)
{
  __shared__ _Float16 Pt[16 * 32];
  __shared__ _Float16 Vl[32 * 96];
  const int lane = threadIdx.x & 31;
  const int ln = lane & 15;
  const int hi = lane >> 4;
  const int qt = blockIdx.x;   // 0..127
  const int h  = blockIdx.y;
  const int b  = blockIdx.z;

  const size_t bh = ((size_t)b * HEADS + h) * SEQ;
  const _Float16* __restrict__ Qp = Qh + bh * HDIM;
  const _Float16* __restrict__ Kp = Kh + bh * HDIM;
  const _Float16* __restrict__ Vp = Vh + bh * HDIM;

  // Q A-fragments for the 3 Hd chunks (16x32 each), live in registers.
  v16h qa[3];
  {
    const int qrow = qt * 16 + ln;
#pragma unroll
    for (int c = 0; c < 3; ++c) {
      const _Float16* qp = Qp + (size_t)qrow * HDIM + c * 32 + hi * 8;
      qa[c] = pack16(*(const v8h*)qp, *(const v8h*)(qp + 16));
    }
  }

  v8f o[6] = {};
  float mrow[8], lrow[8];
#pragma unroll
  for (int r = 0; r < 8; ++r) { mrow[r] = -1e30f; lrow[r] = 0.0f; }

  const float scale = 0.10206207262f;   // 1/sqrt(96)
  const int qhimax  = qt * 16 + 15;
  const int nchunks = qhimax / 32 + 1;  // causal: only chunks with k0 <= qhi

  for (int j = 0; j < nchunks; ++j) {
    const int k0 = j * 32;

    // prefetch next chunk's K/V rows while we compute this one
    if (j + 1 < nchunks) {
      __builtin_prefetch(Kp + (size_t)(k0 + 32 + lane) * HDIM, 0, 0);
      __builtin_prefetch(Vp + (size_t)(k0 + 32 + lane) * HDIM, 0, 0);
    }

    // stage V chunk (rows k0..k0+31, one row per lane) into LDS row-major
    {
      const _Float16* vr = Vp + (size_t)(k0 + lane) * HDIM;
#pragma unroll
      for (int c = 0; c < 12; ++c)
        *(v8h*)&Vl[lane * 96 + c * 8] = *(const v8h*)(vr + c * 8);
    }

    v8f s[2];
#pragma unroll
    for (int t = 0; t < 2; ++t) {
      const int krow = k0 + t * 16 + ln;   // key index for this lane's column
      v8f sc = {};
#pragma unroll
      for (int c = 0; c < 3; ++c) {
        // B = K^T chunk: Hd values {hi*8..+7} and +16 of key row `krow`.
        const _Float16* kp = Kp + (size_t)krow * HDIM + c * 32 + hi * 8;
        v16h kb = pack16(*(const v8h*)kp, *(const v8h*)(kp + 16));
        sc = __builtin_amdgcn_wmma_f32_16x16x32_f16(false, qa[c], false, kb,
                                                    (short)0, sc, false, false);
      }
#pragma unroll
      for (int r = 0; r < 8; ++r) {
        const int qq = qt * 16 + r + hi * 8;
        s[t][r] = (krow <= qq) ? sc[r] * scale : -1e30f;
      }
    }

    // row max over 32 keys: local pairwise max, then xor-reduce within half
    float rmax[8];
#pragma unroll
    for (int r = 0; r < 8; ++r) rmax[r] = fmaxf(s[0][r], s[1][r]);
#pragma unroll
    for (int d = 8; d >= 1; d >>= 1)
#pragma unroll
      for (int r = 0; r < 8; ++r)
        rmax[r] = fmaxf(rmax[r], __shfl_xor(rmax[r], d, 32));

    float alpha[8], rsum[8];
#pragma unroll
    for (int r = 0; r < 8; ++r) {
      const float nm = fmaxf(mrow[r], rmax[r]);
      alpha[r] = __expf(mrow[r] - nm);
      mrow[r] = nm;
      rsum[r] = 0.0f;
    }
#pragma unroll
    for (int t = 0; t < 2; ++t)
#pragma unroll
      for (int r = 0; r < 8; ++r) {
        const float p = __expf(s[t][r] - mrow[r]);
        s[t][r] = p;
        rsum[r] += p;
      }
#pragma unroll
    for (int d = 8; d >= 1; d >>= 1)
#pragma unroll
      for (int r = 0; r < 8; ++r) rsum[r] += __shfl_xor(rsum[r], d, 32);
#pragma unroll
    for (int r = 0; r < 8; ++r) lrow[r] = lrow[r] * alpha[r] + rsum[r];
#pragma unroll
    for (int n = 0; n < 6; ++n)
#pragma unroll
      for (int r = 0; r < 8; ++r) o[n][r] *= alpha[r];

    // Stage P (16x32 f16, row-major) through LDS: C-layout -> A-layout.
#pragma unroll
    for (int t = 0; t < 2; ++t)
#pragma unroll
      for (int r = 0; r < 8; ++r)
        Pt[(r + hi * 8) * 32 + t * 16 + ln] = (_Float16)s[t][r];

    const _Float16* pp = Pt + ln * 32 + hi * 8;
    v16h pa = pack16(*(const v8h*)pp, *(const v8h*)(pp + 16));

    // O += P * V over the 6 Hd column tiles; V B-fragments from LDS.
#pragma unroll
    for (int n = 0; n < 6; ++n) {
      v16h vb;
#pragma unroll
      for (int i = 0; i < 16; ++i)
        vb[i] = Vl[(hi * 8 + i + (i & 8)) * 96 + n * 16 + ln];
      o[n] = __builtin_amdgcn_wmma_f32_16x16x32_f16(false, pa, false, vb,
                                                    (short)0, o[n], false, false);
    }
  }

  float inv[8];
#pragma unroll
  for (int r = 0; r < 8; ++r) inv[r] = 1.0f / lrow[r];
#pragma unroll
  for (int n = 0; n < 6; ++n) {
    const int cc = h * HDIM + n * 16 + ln;   // column in [B,S,D] layout
#pragma unroll
    for (int r = 0; r < 8; ++r) {
      const int srow = qt * 16 + r + hi * 8;
      attn[((size_t)b * SEQ + srow) * EMBED + cc] = (_Float16)(o[n][r] * inv[r]);
    }
  }
}

// ---------------------------------------------------------------------------
// Kernel 3: output projection (f16 attn -> f32 d_out), Wo staged via LDS.
// ---------------------------------------------------------------------------
__global__ __launch_bounds__(128, 1)
void out_proj_kernel(const _Float16* __restrict__ attn,
                     const float* __restrict__ Wo, const float* __restrict__ bo,
                     float* __restrict__ out)
{
  __shared__ _Float16 Wl[16 * 32];   // [col][k] transposed chunk
  const int lane = threadIdx.x & 31;
  const int wave = threadIdx.x >> 5;
  const int ln = lane & 15;
  const int hi = lane >> 4;
  const int tileM = blockIdx.x * 4 + wave;
  const int tileN = blockIdx.y;
  const int arow = tileM * 16 + ln;
  const int ccol = tileN * 16 + ln;

  const int scl = threadIdx.x & 15;
  const int sk0 = threadIdx.x >> 4;

  v8f acc = {};
#pragma unroll 1
  for (int kk = 0; kk < EMBED; kk += 32) {
    __syncthreads();
#pragma unroll
    for (int j = 0; j < 4; ++j) {
      const int kl = sk0 + 8 * j;
      Wl[scl * 32 + kl] =
          (_Float16)Wo[(size_t)(kk + kl) * EMBED + tileN * 16 + scl];
    }
    __syncthreads();

    const _Float16* ap = attn + (size_t)arow * EMBED + kk + hi * 8;
    v16h a = pack16(*(const v8h*)ap, *(const v8h*)(ap + 16));
    const _Float16* wp = &Wl[ln * 32 + hi * 8];
    v16h bfrag = pack16(*(const v8h*)wp, *(const v8h*)(wp + 16));
    acc = __builtin_amdgcn_wmma_f32_16x16x32_f16(false, a, false, bfrag,
                                                 (short)0, acc, false, false);
  }
  const float bc = bo[ccol];
  float* op = out + (size_t)(tileM * 16 + hi * 8) * EMBED + ccol;
#pragma unroll
  for (int r = 0; r < 8; ++r)
    op[(size_t)r * EMBED] = acc[r] + bc;
}

extern "C" void kernel_launch(void* const* d_in, const int* in_sizes, int n_in,
                              void* d_out, int out_size, void* d_ws, size_t ws_size,
                              hipStream_t stream) {
  (void)in_sizes; (void)n_in; (void)out_size; (void)ws_size;
  const float* x  = (const float*)d_in[0];
  const float* Wq = (const float*)d_in[1];
  const float* bq = (const float*)d_in[2];
  const float* Wk = (const float*)d_in[3];
  const float* bk = (const float*)d_in[4];
  const float* Wv = (const float*)d_in[5];
  const float* bv = (const float*)d_in[6];
  const float* Wo = (const float*)d_in[7];
  const float* bo = (const float*)d_in[8];
  float* out = (float*)d_out;

  const size_t N1 = (size_t)BATCH * HEADS * SEQ * HDIM;  // 6,291,456 elems
  _Float16* Qh   = (_Float16*)d_ws;
  _Float16* Kh   = Qh + N1;
  _Float16* Vh   = Kh + N1;
  _Float16* attn = Vh + N1;   // total ws use: 4*N1*2 = 50.3 MB

  dim3 gqkv(NROWS / 64, EMBED / 16, 3);     // (256, 24, 3), 4 waves/block
  qkv_proj_kernel<<<gqkv, 128, 0, stream>>>(x, Wq, bq, Wk, bk, Wv, bv,
                                            Qh, Kh, Vh);

  dim3 gfa(SEQ / 16, HEADS, BATCH);         // (128, 4, 8), 1 wave/block
  flash_attn_kernel<<<gfa, 32, 0, stream>>>(Qh, Kh, Vh, attn);

  dim3 gout(NROWS / 64, EMBED / 16);        // (256, 24)
  out_proj_kernel<<<gout, 128, 0, stream>>>(attn, Wo, bo, out);
}
